// TopKRoutingBiasedSAE_56745107915434
// MI455X (gfx1250) — matrix operand order repaired and checked by reading, not compiled, not run on hardware
//
#include <hip/hip_runtime.h>
#include <hip/hip_bf16.h>

#define DIM   2048
#define HID   16384
#define NROW  8192
#define TOPK  32

#define RB    32      // rows per workgroup
#define HT    128     // H columns per tile (8 waves x 16)
#define KC    64      // K elements staged per LDS stage (2 wmma k-steps)
#define NSTG  (DIM / KC)            // 32 stages per H tile
#define NST   ((HID / HT) * NSTG)   // 4096 total pipeline stages
#define NBUF  4       // LDS stage buffers (3 stages in flight)
#define CAND  128     // candidate buffer per row per tile

typedef __attribute__((ext_vector_type(16))) __bf16 v16bf;
typedef __attribute__((ext_vector_type(8)))  float  v8f;
typedef __attribute__((ext_vector_type(4)))  int    v4i_;

// ---- CDNA5 async global->LDS staging (ASYNCcnt path), with safe fallback ----
#if defined(__HIP_DEVICE_COMPILE__) && __has_builtin(__builtin_amdgcn_global_load_async_to_lds_b128)
#define HAVE_ASYNC 1
#else
#define HAVE_ASYNC 0
#endif

__device__ __forceinline__ void async_copy16(void* lds, const void* g) {
#if HAVE_ASYNC
  __builtin_amdgcn_global_load_async_to_lds_b128(
      (__attribute__((address_space(1))) v4i_*)(g),
      (__attribute__((address_space(3))) v4i_*)(lds), /*offset=*/0, /*cpol=*/0);
#else
  *(uint4*)lds = *(const uint4*)g;
#endif
}

template <int N>
__device__ __forceinline__ void wait_async() {
#if HAVE_ASYNC
#if __has_builtin(__builtin_amdgcn_s_wait_asynccnt)
  __builtin_amdgcn_s_wait_asynccnt(N);
#else
  asm volatile("s_wait_asynccnt %0" ::"i"(N) : "memory");
#endif
#endif
}

// ---------------- K0a: f32 -> bf16 bulk convert (W_enc) ----------------
__global__ __launch_bounds__(256) void k_cvt_bf16(const float* __restrict__ in,
                                                  __bf16* __restrict__ out) {
  size_t i = ((size_t)blockIdx.x * 256 + threadIdx.x) * 8;
  float4 f0 = *(const float4*)(in + i);
  float4 f1 = *(const float4*)(in + i + 4);
  union { uint4 q; __bf16 h[8]; } r;
  r.h[0] = (__bf16)f0.x; r.h[1] = (__bf16)f0.y; r.h[2] = (__bf16)f0.z; r.h[3] = (__bf16)f0.w;
  r.h[4] = (__bf16)f1.x; r.h[5] = (__bf16)f1.y; r.h[6] = (__bf16)f1.z; r.h[7] = (__bf16)f1.w;
  *(uint4*)(out + i) = r.q;
}

// ---------------- K0b: xc = bf16(x - b_dec) ----------------
__global__ __launch_bounds__(256) void k_center_x(const float* __restrict__ x,
                                                  const float* __restrict__ bdec,
                                                  __bf16* __restrict__ out) {
  size_t i = ((size_t)blockIdx.x * 256 + threadIdx.x) * 8;
  int d = (int)(i & (size_t)(DIM - 1));
  float4 f0 = *(const float4*)(x + i);
  float4 f1 = *(const float4*)(x + i + 4);
  float4 b0 = *(const float4*)(bdec + d);
  float4 b1 = *(const float4*)(bdec + d + 4);
  union { uint4 q; __bf16 h[8]; } r;
  r.h[0] = (__bf16)(f0.x - b0.x); r.h[1] = (__bf16)(f0.y - b0.y);
  r.h[2] = (__bf16)(f0.z - b0.z); r.h[3] = (__bf16)(f0.w - b0.w);
  r.h[4] = (__bf16)(f1.x - b1.x); r.h[5] = (__bf16)(f1.y - b1.y);
  r.h[6] = (__bf16)(f1.z - b1.z); r.h[7] = (__bf16)(f1.w - b1.w);
  *(uint4*)(out + i) = r.q;
}

// ---------------- K0c: W_dec [D,H] f32 -> W_decT [H,D] bf16 ----------------
__global__ __launch_bounds__(256) void k_transpose_wdec(const float* __restrict__ wdec,
                                                        __bf16* __restrict__ outT) {
  __shared__ float tile[32][33];
  const int tx = threadIdx.x & 31;
  const int ty = threadIdx.x >> 5;          // 0..7
  const int hb = blockIdx.x * 32;
  const int db = blockIdx.y * 32;
#pragma unroll
  for (int yy = 0; yy < 4; ++yy) {
    int d = db + ty * 4 + yy;
    tile[ty * 4 + yy][tx] = wdec[(size_t)d * HID + hb + tx];
  }
  __syncthreads();
#pragma unroll
  for (int yy = 0; yy < 4; ++yy) {
    int h = hb + ty * 4 + yy;
    outT[(size_t)h * DIM + db + tx] = (__bf16)tile[tx][ty * 4 + yy];
  }
}

// ---------------- K1: fused encode GEMM + streaming top-32 + sparse decode ----------------
struct Smem {
  __bf16 xt[RB][DIM + 8];         // 131,584 B : resident xc rows (bank-padded)
  __bf16 wt[NBUF][HT][KC + 8];    //  73,728 B : 4-deep W_enc stage tiles
  float  topv[RB][TOPK];
  int    topi[RB][TOPK];
  float  candv[RB][CAND];
  int    candi[RB][CAND];
  float  thr[RB];
  int    cnt[RB];
  int    ccnt[RB];
};

__global__ __launch_bounds__(256) void k_sae(const __bf16* __restrict__ xc,
                                             const __bf16* __restrict__ wenc,
                                             const float*  __restrict__ benc,
                                             const __bf16* __restrict__ wdecT,
                                             const float*  __restrict__ bdec,
                                             float* __restrict__ out) {
  extern __shared__ char smem_raw[];
  Smem& S = *(Smem*)smem_raw;

  const int tid  = threadIdx.x;
  const int wave = tid >> 5;        // 0..7 -> owns 16 output cols
  const int lane = tid & 31;
  const int l16  = lane & 15;
  const bool hi  = lane >= 16;
  const int row0 = blockIdx.x * RB;

  if (tid < RB) { S.cnt[tid] = 0; S.ccnt[tid] = 0; S.thr[tid] = -3.4e38f; }

  // issue one pipeline stage (4 async b128 per thread) into LDS buffer `buf`
  auto issue_wt = [&](int g, int buf) {
    const int hb = (g >> 5) * HT;              // H tile base
    const int st = g & (NSTG - 1);             // K stage within tile
    const __bf16* gb = wenc + (size_t)hb * DIM + st * KC;
#pragma unroll
    for (int i = tid; i < HT * KC / 8; i += 256) {
      int r = (i * 8) >> 6;
      int c = (i * 8) & (KC - 1);
      async_copy16(&S.wt[buf][r][c], gb + (size_t)r * DIM + c);
    }
  };

  // Stage the 32 centered bf16 rows (full K) into LDS once (async engine).
  for (int i = tid; i < RB * DIM / 8; i += 256) {
    int r = (i * 8) >> 11;
    int c = (i * 8) & (DIM - 1);
    async_copy16(&S.xt[r][c], xc + (size_t)(row0 + r) * DIM + c);
  }
  issue_wt(0, 0);   // prime 2 stages
  issue_wt(1, 1);

  v8f acc0 = {0.f, 0.f, 0.f, 0.f, 0.f, 0.f, 0.f, 0.f};
  v8f acc1 = acc0;

  for (int g = 0; g < NST; ++g) {
    if (g + 2 < NST) issue_wt(g + 2, (g + 2) & (NBUF - 1));
    // Retire everything older than the two newest stages (ASYNCcnt is in-order):
    // guarantees stage g (and initial x staging) has landed in LDS.
    if (g + 3 <= NST)      wait_async<8>();
    else if (g + 2 == NST) wait_async<4>();
    else                   wait_async<0>();
    __syncthreads();

    const int buf = g & (NBUF - 1);
    const int st  = g & (NSTG - 1);

#pragma unroll
    for (int j = 0; j < 2; ++j) {
      const int kb = j * 32;
      union Frag { v16bf v; uint4 q[2]; };
      // B fragment: lane<16 -> col l16, K kb..kb+15 ; lane>=16 -> K kb+16..kb+31
      Frag b;
      const __bf16* bp = &S.wt[buf][wave * 16 + l16][kb + (hi ? 16 : 0)];
      b.q[0] = *(const uint4*)bp;
      b.q[1] = *(const uint4*)(bp + 8);
      // A fragments (row tiles 0/1): lane<16 -> K +0..15 ; lane>=16 -> +16..31
      const int ka = st * KC + kb + (hi ? 16 : 0);
      Frag a0, a1;
      const __bf16* ap0 = &S.xt[l16][ka];
      a0.q[0] = *(const uint4*)ap0;
      a0.q[1] = *(const uint4*)(ap0 + 8);
      const __bf16* ap1 = &S.xt[16 + l16][ka];
      a1.q[0] = *(const uint4*)ap1;
      a1.q[1] = *(const uint4*)(ap1 + 8);

      acc0 = __builtin_amdgcn_wmma_f32_16x16x32_bf16(false, a0.v, false, b.v,
                                                     (short)0, acc0, false, false);
      acc1 = __builtin_amdgcn_wmma_f32_16x16x32_bf16(false, a1.v, false, b.v,
                                                     (short)0, acc1, false, false);
    }

    if (st == NSTG - 1) {
      // ---- streaming top-k: threshold-pruned candidate append + merge ----
      const int hbase = (g >> 5) * HT;
      const int hcol  = hbase + wave * 16 + l16;
      const float be  = benc[hcol];
#pragma unroll
      for (int t = 0; t < 2; ++t) {
#pragma unroll
        for (int j = 0; j < 8; ++j) {
          int row = t * 16 + (hi ? j + 8 : j);
          float v = (t == 0 ? acc0[j] : acc1[j]) + be;
          if (S.cnt[row] < TOPK || v > S.thr[row]) {
            int p = atomicAdd(&S.ccnt[row], 1);
            if (p < CAND) { S.candv[row][p] = v; S.candi[row][p] = hcol; }
          }
        }
      }
      __syncthreads();
      if (tid < RB) {           // one thread merges its row's candidates
        const int row = tid;
        int n = S.ccnt[row]; if (n > CAND) n = CAND;
        int c = S.cnt[row];
        float th = S.thr[row];
        for (int q = 0; q < n; ++q) {
          float v = S.candv[row][q];
          int   h = S.candi[row][q];
          if (c < TOPK) {
            S.topv[row][c] = v; S.topi[row][c] = h; ++c;
            if (c == TOPK) {
              float m = S.topv[row][0];
              for (int s = 1; s < TOPK; ++s) m = fminf(m, S.topv[row][s]);
              th = m;
            }
          } else if (v > th) {
            int mi = 0; float mv = S.topv[row][0];
            for (int s = 1; s < TOPK; ++s)
              if (S.topv[row][s] < mv) { mv = S.topv[row][s]; mi = s; }
            S.topv[row][mi] = v; S.topi[row][mi] = h;
            mv = S.topv[row][0];
            for (int s = 1; s < TOPK; ++s) mv = fminf(mv, S.topv[row][s]);
            th = mv;
          }
        }
        S.cnt[row] = c; S.thr[row] = th; S.ccnt[row] = 0;
      }
      acc0 = (v8f){0.f, 0.f, 0.f, 0.f, 0.f, 0.f, 0.f, 0.f};
      acc1 = acc0;
    }
  }
  __syncthreads();

  // ---- sparse decode: out[row,:] = b_dec + sum_k relu(topv)*W_decT[topi,:] ----
  const int d0 = tid * 8;
  for (int r = 0; r < RB; ++r) {
    float4 b0 = *(const float4*)(bdec + d0);
    float4 b1 = *(const float4*)(bdec + d0 + 4);
    float o[8] = {b0.x, b0.y, b0.z, b0.w, b1.x, b1.y, b1.z, b1.w};
    for (int k = 0; k < TOPK; ++k) {
      float a = S.topv[r][k];
      a = a > 0.f ? a : 0.f;
      int h = S.topi[r][k];
      union { uint4 q; __bf16 w[8]; } wv;
      wv.q = *(const uint4*)(wdecT + (size_t)h * DIM + d0);
#pragma unroll
      for (int i = 0; i < 8; ++i) o[i] = fmaf(a, (float)wv.w[i], o[i]);
    }
    float4 o0 = {o[0], o[1], o[2], o[3]};
    float4 o1 = {o[4], o[5], o[6], o[7]};
    float* op = out + (size_t)(row0 + r) * DIM + d0;
    *(float4*)op = o0;
    *(float4*)(op + 4) = o1;
  }
}

extern "C" void kernel_launch(void* const* d_in, const int* in_sizes, int n_in,
                              void* d_out, int out_size, void* d_ws, size_t ws_size,
                              hipStream_t stream) {
  const float* x     = (const float*)d_in[0];
  const float* W_enc = (const float*)d_in[1];
  const float* b_enc = (const float*)d_in[2];
  const float* W_dec = (const float*)d_in[3];
  const float* b_dec = (const float*)d_in[4];
  float* out = (float*)d_out;

  char* ws = (char*)d_ws;
  __bf16* xws    = (__bf16*)ws;                              // 32 MB: bf16(x - b_dec)
  __bf16* wencws = (__bf16*)(ws + ((size_t)32 << 20));       // 64 MB: bf16 W_enc
  __bf16* wdecT  = (__bf16*)(ws + ((size_t)96 << 20));       // 64 MB: bf16 W_dec^T

  k_center_x<<<NROW * DIM / (256 * 8), 256, 0, stream>>>(x, b_dec, xws);
  k_cvt_bf16<<<HID * DIM / (256 * 8), 256, 0, stream>>>(W_enc, wencws);
  dim3 tg(HID / 32, DIM / 32);
  k_transpose_wdec<<<tg, 256, 0, stream>>>(W_dec, wdecT);
  k_sae<<<NROW / RB, 256, sizeof(Smem), stream>>>(xws, wencws, b_enc, wdecT, b_dec, out);
}
